// SphereConv_80496277061854
// MI455X (gfx1250) — matrix operand order repaired and checked by reading, not compiled, not run on hardware
//
#include <hip/hip_runtime.h>
#include <hip/hip_bf16.h>

// Problem constants (from reference): N=2, CIN=COUT=64, H=256, W=512, K=3 (9 taps)
#define HH   256
#define WW   512
#define CI   64
#define CO   64
#define K2   9
#define TP   64     // pixels per block tile (along W)
#define SROW 72     // padded LDS row stride in f16 units (144 B, 16B-aligned rows)

typedef __attribute__((ext_vector_type(16))) _Float16 v16h;
typedef __attribute__((ext_vector_type(8)))  _Float16 v8h;
typedef __attribute__((ext_vector_type(8)))  float    v8f;

union V16 { v16h v; v8h h[2]; };

// ---------------------------------------------------------------------------
// Pre-kernel: repack weight[o][c][kh][kw] (f32) -> Wh[k][o][c] (f16) in d_ws.
// This layout makes each WMMA A-fragment two aligned 16B global loads per lane.
// ---------------------------------------------------------------------------
__global__ void SphereConv_weights_f16_kernel(const float* __restrict__ w,
                                              _Float16* __restrict__ wh) {
    int idx = blockIdx.x * blockDim.x + threadIdx.x;   // over K2*CO*CI = 36864
    if (idx >= K2 * CO * CI) return;
    int c = idx & 63;
    int o = (idx >> 6) & 63;
    int k = idx >> 12;
    wh[idx] = (_Float16)w[(o * CI + c) * K2 + k];
}

// ---------------------------------------------------------------------------
// Main kernel: gather-interp into LDS (f16), then WMMA f16 x f16 -> f32.
// Block = 256 threads (8 wave32). Tile = 1 batch x 1 row x 64 pixels.
// ---------------------------------------------------------------------------
__global__ __launch_bounds__(256) void SphereConv_wmma_kernel(
    const float*     __restrict__ x,     // [N][CI][H][W] f32
    const _Float16*  __restrict__ wh,    // [K2][CO][CI]  f16
    const float*     __restrict__ pos,   // [K2*2][H][W]  f32 (lat,lon interleaved per tap)
    float*           __restrict__ out)   // [N][CO][H][W] f32
{
    __shared__ __align__(16) _Float16 slds[TP * SROW];   // 9216 B sampled tile S[pixel][c]

    const int bx = blockIdx.x;
    const int wt = bx & 7;                 // W-tile (8 tiles of 64)
    const int h  = (bx >> 3) & (HH - 1);   // row
    const int n  = bx >> 11;               // batch

    const int t    = threadIdx.x;
    const int lane = t & 31;
    const int wave = t >> 5;

    // --- sampling-phase mapping: thread -> (pixel, 16-channel group) ---
    const int sp   = t & 63;               // pixel in tile
    const int cg   = t >> 6;               // channel group 0..3 (x16)
    const int wcol = wt * TP + sp;

    // --- WMMA-phase mapping: wave -> (pixel-tile, pair of COUT-tiles) ---
    const int ptile = wave & 3;            // 16-pixel tile 0..3
    const int ot0   = (wave >> 2) * 2;     // COUT tiles {ot0, ot0+1}
    const int ln    = lane & 15;
    const int half  = lane >> 4;

    v8f acc0 = {};                          // D-tile: COUT tile ot0
    v8f acc1 = {};                          // D-tile: COUT tile ot0+1

    const float* posp = pos + h * WW + wcol;             // + tap*2*H*W strides
    const float* xn   = x + (size_t)n * CI * HH * WW;

    for (int k = 0; k < K2; ++k) {
        // ================= bilinear sampling -> LDS (f16) =================
        float py = posp[(2 * k + 0) * (HH * WW)];
        float px = posp[(2 * k + 1) * (HH * WW)];
        float y0f = floorf(py), x0f = floorf(px);
        float dy = py - y0f,    dx = px - x0f;
        // reference semantics: clip index, zero the corner weight if outside
        float vy0 = (y0f >=  0.f && y0f <= (float)(HH - 1)) ? 1.f : 0.f;
        float vy1 = (y0f >= -1.f && y0f <= (float)(HH - 2)) ? 1.f : 0.f;
        float vx0 = (x0f >=  0.f && x0f <= (float)(WW - 1)) ? 1.f : 0.f;
        float vx1 = (x0f >= -1.f && x0f <= (float)(WW - 2)) ? 1.f : 0.f;
        float w00 = (1.f - dy) * (1.f - dx) * vy0 * vx0;
        float w01 = (1.f - dy) * dx         * vy0 * vx1;
        float w10 = dy         * (1.f - dx) * vy1 * vx0;
        float w11 = dy         * dx         * vy1 * vx1;
        int iy0 = min(max((int)y0f,     0), HH - 1);
        int iy1 = min(max((int)y0f + 1, 0), HH - 1);
        int ix0 = min(max((int)x0f,     0), WW - 1);
        int ix1 = min(max((int)x0f + 1, 0), WW - 1);
        int o00 = iy0 * WW + ix0, o01 = iy0 * WW + ix1;
        int o10 = iy1 * WW + ix0, o11 = iy1 * WW + ix1;

        #pragma unroll
        for (int i = 0; i < 16; ++i) {
            const float* xp = xn + (size_t)(cg * 16 + i) * (HH * WW);
            float v = w00 * xp[o00] + w01 * xp[o01] + w10 * xp[o10] + w11 * xp[o11];
            slds[sp * SROW + cg * 16 + i] = (_Float16)v;
        }
        __syncthreads();

        // ================= WMMA: two K=32 chunks over CIN=64 ==============
        #pragma unroll
        for (int ch = 0; ch < 2; ++ch) {
            // B fragment (32x16): lane = column(pixel), K runs {0..7,16..23}/{8..15,24..31}
            V16 bf;
            const _Float16* bp = &slds[(ptile * 16 + ln) * SROW + ch * 32 + half * 8];
            bf.h[0] = *(const v8h*)bp;          // K = base .. base+7
            bf.h[1] = *(const v8h*)(bp + 16);   // K = base+16 .. base+23

            // A fragments (16x32): lane = row(COUT); K runs {0..7,16..23}/{8..15,24..31}
            const int ca = ch * 32 + half * 8;
            V16 a0, a1;
            const _Float16* ap0 = wh + ((k * CO + ot0 * 16 + ln) * CI + ca);
            a0.h[0] = *(const v8h*)ap0;
            a0.h[1] = *(const v8h*)(ap0 + 16);
            const _Float16* ap1 = ap0 + 16 * CI;
            a1.h[0] = *(const v8h*)ap1;
            a1.h[1] = *(const v8h*)(ap1 + 16);

            acc0 = __builtin_amdgcn_wmma_f32_16x16x32_f16(
                       false, a0.v, false, bf.v, (short)0, acc0, false, false);
            acc1 = __builtin_amdgcn_wmma_f32_16x16x32_f16(
                       false, a1.v, false, bf.v, (short)0, acc1, false, false);
        }
        __syncthreads();
    }

    // ================= epilogue: C/D layout -> NCHW f32 ===================
    const int wbase = wt * TP + ptile * 16 + ln;
    #pragma unroll
    for (int j = 0; j < 8; ++j) {
        int m = j + 8 * half;  // row within 16x16 D-tile
        out[(((size_t)n * CO + (ot0 + 0) * 16 + m) * HH + h) * WW + wbase] = acc0[j];
        out[(((size_t)n * CO + (ot0 + 1) * 16 + m) * HH + h) * WW + wbase] = acc1[j];
    }
}

// ---------------------------------------------------------------------------
extern "C" void kernel_launch(void* const* d_in, const int* in_sizes, int n_in,
                              void* d_out, int out_size, void* d_ws, size_t ws_size,
                              hipStream_t stream) {
    const float* x      = (const float*)d_in[0];   // (2,64,256,512) f32
    const float* weight = (const float*)d_in[1];   // (64,64,3,3)    f32
    const float* pos    = (const float*)d_in[2];   // (18,256,512)   f32
    float*       out    = (float*)d_out;           // (2,64,256,512) f32
    _Float16*    wh     = (_Float16*)d_ws;         // needs 9*64*64*2 = 73728 B

    // 1) repack + downconvert weights: [o][c][k] f32 -> [k][o][c] f16
    SphereConv_weights_f16_kernel<<<dim3((K2 * CO * CI + 255) / 256), dim3(256), 0, stream>>>(
        weight, wh);

    // 2) fused gather-interp + WMMA GEMM
    const int nblocks = 2 * HH * (WW / TP);        // 4096
    SphereConv_wmma_kernel<<<dim3(nblocks), dim3(256), 0, stream>>>(x, wh, pos, out);
}
